// MOORENetwork_83571473646283
// MI455X (gfx1250) — matrix operand order repaired and checked by reading, not compiled, not run on hardware
//
#include <hip/hip_runtime.h>
#include <math.h>

#define B_SZ   4096
#define OBS    512
#define T_TASK 10
#define E_EXP  8
#define W_DIM  1024
#define H_DIM  8
#define XW     (OBS + T_TASK)
#define EPS_F  1e-8f

// LDS pitches (halves) — 16B row padding for bank-conflict-free ds_load_b128
#define APITCH 40    // 32 payload + 8 pad  (row = 80B,  20 words)
#define BPITCH 72    // 64 payload + 8 pad  (row = 144B, 36 words)

typedef __attribute__((ext_vector_type(16))) _Float16 v16h;
typedef __attribute__((ext_vector_type(8)))  _Float16 v8h;
typedef __attribute__((ext_vector_type(8)))  float    v8f;
typedef __attribute__((ext_vector_type(4)))  unsigned int v4u;
typedef __attribute__((ext_vector_type(8)))  int      v8i;
typedef __attribute__((ext_vector_type(4)))  int      v4i;

#if defined(__has_builtin)
#if __has_builtin(__builtin_amdgcn_tensor_load_to_lds) && __has_builtin(__builtin_amdgcn_s_wait_tensorcnt)
#define HAVE_TDM 1
#endif
#endif

// ---------------------------------------------------------------------------
// Pack / convert kernels
// ---------------------------------------------------------------------------
__global__ __launch_bounds__(256) void pack_feat_kernel(const float* __restrict__ x,
                                                        _Float16* __restrict__ dst) {
    int i = blockIdx.x * blockDim.x + threadIdx.x;          // over B*OBS
    if (i < B_SZ * OBS) {
        int b = i / OBS, d = i - b * OBS;
        dst[i] = (_Float16)x[(size_t)b * XW + d];
    }
}

__global__ __launch_bounds__(256) void f32_to_f16_kernel(const float* __restrict__ src,
                                                         _Float16* __restrict__ dst, int n) {
    int i = blockIdx.x * blockDim.x + threadIdx.x;
    if (i < n) dst[i] = (_Float16)src[i];
}

// task_emb[b][e] = sum_t onehot[b][t] * W_task[t][e]; ti[b] = argmax_t onehot
__global__ __launch_bounds__(256) void task_kernel(const float* __restrict__ x,
                                                   const float* __restrict__ W_task,
                                                   float* __restrict__ taskEmb,
                                                   int* __restrict__ ti) {
    int b = blockIdx.x * blockDim.x + threadIdx.x;
    if (b >= B_SZ) return;
    const float* oh = x + (size_t)b * XW + OBS;
    float best = -3.402823466e+38f;
    int bi = 0;
    float emb[E_EXP];
#pragma unroll
    for (int e = 0; e < E_EXP; ++e) emb[e] = 0.0f;
    for (int t = 0; t < T_TASK; ++t) {
        float v = oh[t];
        if (v > best) { best = v; bi = t; }
#pragma unroll
        for (int e = 0; e < E_EXP; ++e) emb[e] += v * W_task[t * E_EXP + e];
    }
#pragma unroll
    for (int e = 0; e < E_EXP; ++e) taskEmb[(size_t)b * E_EXP + e] = emb[e];
    ti[b] = bi;
}

#ifdef HAVE_TDM
// ---------------------------------------------------------------------------
// TDM: issue a 2D tensor_load_to_lds (tile_y rows x tile_x f16 elements).
//   D# group0: count=1 | lds_addr | global_addr[56:0] | type=2
//   D# group1: data_size=1 (2B), optional LDS row padding, dims/strides
// pad_interval code k -> pad every 2^(k+1) DWORDs; pad_amount code c -> c+1 DWORDs
// This toolchain's builtin takes 6 operands (g0, g1, g2, g3, g4, cpol);
// the trailing descriptor words are zero for a 2D tensor.
// ---------------------------------------------------------------------------
__device__ __forceinline__ void tdm_load_2d(const void* gaddr, unsigned lds_off,
                                            unsigned tile_x, unsigned tile_y,
                                            unsigned stride_elems,
                                            unsigned pad_interval, unsigned pad_amount) {
    unsigned long long ga = (unsigned long long)(uintptr_t)gaddr;
    v4u g0;
    g0[0] = 1u;                                              // count=1, user mode
    g0[1] = lds_off;                                         // lds_addr (bytes)
    g0[2] = (unsigned)ga;                                    // global_addr[31:0]
    g0[3] = (unsigned)((ga >> 32) & 0x01FFFFFFu) | (2u << 30); // addr[56:32] | type=2
    v8i g1;
    g1[0] = (int)((1u << 16) |                               // data_size = 2 bytes
                  (1u << 20) |                               // pad_enable
                  (pad_interval << 22) | (pad_amount << 25));
    unsigned tdim0 = stride_elems;                           // >= tile_x, tile stays in-bounds
    unsigned tdim1 = 1u << 20;
    g1[1] = (int)((tdim0 & 0xFFFFu) << 16);                  // tensor_dim0[15:0] @ bits 63:48
    g1[2] = (int)(((tdim0 >> 16) & 0xFFFFu) | ((tdim1 & 0xFFFFu) << 16));
    g1[3] = (int)(((tdim1 >> 16) & 0xFFFFu) | (tile_x << 16)); // tile_dim0 @ bits 127:112
    g1[4] = (int)tile_y;                                     // tile_dim1 @ bits 143:128
    g1[5] = (int)stride_elems;                               // tensor_dim0_stride[31:0]
    g1[6] = 0;
    g1[7] = 0;
    v4i g2 = {0, 0, 0, 0};
    v4i g3 = {0, 0, 0, 0};
    v8i g4 = {0, 0, 0, 0, 0, 0, 0, 0};
    __builtin_amdgcn_tensor_load_to_lds(g0, g1, g2, g3, g4, 0);
}
#endif

// ---------------------------------------------------------------------------
// Batched per-expert WMMA GEMM: Out[b][e][n] = act(A * Bw[e] + bias[e])
// Block = 256 threads (8 waves). Block tile: M=128 (wave = 16 rows), N=64.
// TDM double-buffers A (128x32) and B (32x64) slabs in LDS; waves run
// ds_load_b128 + v_wmma_f32_16x16x32_f16 only. Grid: (N/64, M/128, E).
// ---------------------------------------------------------------------------
template <bool RELU, typename OutT>
__global__ __launch_bounds__(256) void moe_gemm_kernel(
    const _Float16* __restrict__ A, int aExpOff, int aStride,
    const _Float16* __restrict__ Bw,
    const float* __restrict__ bias,
    OutT* __restrict__ Out, int K) {
    const int N    = W_DIM;
    const int lane = threadIdx.x & 31;
    const int wv   = threadIdx.x >> 5;
    const int e    = blockIdx.z;
    const int m0   = blockIdx.y * 128 + wv * 16;
    const int n0   = blockIdx.x * 64;
    const int lo   = lane & 15, hi = lane >> 4;

    const float* be = bias + (size_t)e * N;
    OutT*        Oe = Out + (size_t)e * N;

    v8f acc[4] = {};

#ifdef HAVE_TDM
    __shared__ _Float16 sA[2][128 * APITCH];
    __shared__ _Float16 sB[2][32 * BPITCH];

    const _Float16* Abase = A + (size_t)e * aExpOff + (size_t)(blockIdx.y * 128) * aStride;
    const _Float16* Bbase = Bw + (size_t)e * (size_t)K * N + n0;

    const unsigned ldsA[2] = {(unsigned)(uintptr_t)&sA[0][0], (unsigned)(uintptr_t)&sA[1][0]};
    const unsigned ldsB[2] = {(unsigned)(uintptr_t)&sB[0][0], (unsigned)(uintptr_t)&sB[1][0]};

    const int nk = K >> 5;
    if (wv == 0) {
        // A rows 64B payload: pad every 16 DWORDs (code 3) by 4 DWORDs (code 3)
        tdm_load_2d(Abase, ldsA[0], 32, 128, (unsigned)aStride, 3u, 3u);
        // B rows 128B payload: pad every 32 DWORDs (code 4) by 4 DWORDs (code 3)
        tdm_load_2d(Bbase, ldsB[0], 64, 32, (unsigned)N, 4u, 3u);
    }

    for (int kb = 0; kb < nk; ++kb) {
        const int p = kb & 1;
        if (wv == 0) __builtin_amdgcn_s_wait_tensorcnt(0);
        __syncthreads();                                  // buffer p ready for all waves
        if (wv == 0 && kb + 1 < nk) {                     // prefetch next slab into p^1
            tdm_load_2d(Abase + (size_t)(kb + 1) * 32, ldsA[p ^ 1], 32, 128,
                        (unsigned)aStride, 3u, 3u);
            tdm_load_2d(Bbase + (size_t)(kb + 1) * 32 * N, ldsB[p ^ 1], 64, 32,
                        (unsigned)N, 4u, 3u);
        }

        // ---- A fragment (16x32): lane (lo,hi) -> row wv*16+lo,
        //      halves hi*8+{0..7} and hi*8+{16..23}
        const v8h* ap = (const v8h*)&sA[p][(wv * 16 + lo) * APITCH + hi * 8];
        v8h alow  = ap[0];
        v8h ahigh = ap[2];
        v16h a = __builtin_shufflevector(alow, ahigh,
                                         0, 1, 2, 3, 4, 5, 6, 7,
                                         8, 9, 10, 11, 12, 13, 14, 15);

        // ---- B fragments (32x16 each): lane holds column N=lo, K = hi*8+{0..7}, 16+hi*8+{0..7}
        const _Float16* bp = &sB[p][lo];
        v16h bfrag[4];
#pragma unroll
        for (int nt = 0; nt < 4; ++nt) {
#pragma unroll
            for (int s = 0; s < 16; ++s) {
                const int kk = (s < 8) ? (hi * 8 + s) : (16 + hi * 8 + (s - 8));
                bfrag[nt][s] = bp[kk * BPITCH + nt * 16];
            }
        }
#pragma unroll
        for (int nt = 0; nt < 4; ++nt) {
            acc[nt] = __builtin_amdgcn_wmma_f32_16x16x32_f16(
                false, a, false, bfrag[nt], (short)0, acc[nt], false, false);
        }
        __syncthreads();                                  // reads of buffer p done
    }
#else
    // Fallback: direct global-load pipeline
    const _Float16* Ae = A + (size_t)e * aExpOff + (size_t)(m0 + lo) * aStride + hi * 8;
    const _Float16* Be = Bw + (size_t)e * (size_t)K * N + n0 + lo;
    for (int k0 = 0; k0 < K; k0 += 32) {
        const v8h* av  = (const v8h*)(Ae + k0);
        v8h alow  = av[0];
        v8h ahigh = av[2];
        v16h a = __builtin_shufflevector(alow, ahigh,
                                         0, 1, 2, 3, 4, 5, 6, 7,
                                         8, 9, 10, 11, 12, 13, 14, 15);
        const _Float16* bv = Be + (size_t)k0 * N;
        v16h bfrag[4];
#pragma unroll
        for (int nt = 0; nt < 4; ++nt) {
#pragma unroll
            for (int s = 0; s < 16; ++s) {
                const int kk = (s < 8) ? (hi * 8 + s) : (16 + hi * 8 + (s - 8));
                bfrag[nt][s] = bv[(size_t)kk * N + nt * 16];
            }
        }
#pragma unroll
        for (int nt = 0; nt < 4; ++nt) {
            acc[nt] = __builtin_amdgcn_wmma_f32_16x16x32_f16(
                false, a, false, bfrag[nt], (short)0, acc[nt], false, false);
        }
    }
#endif

    // Epilogue: D layout — lane (lo,hi), VGPR r -> (M = r + 8*hi, N = lo)
#pragma unroll
    for (int nt = 0; nt < 4; ++nt) {
        const int n    = n0 + nt * 16 + lo;
        const float bv = be[n];
#pragma unroll
        for (int r = 0; r < 8; ++r) {
            float v = acc[nt][r] + bv;
            if (RELU) v = fmaxf(v, 0.0f);
            Oe[(size_t)(m0 + r + 8 * hi) * (E_EXP * W_DIM) + n] = (OutT)v;
        }
    }
}

// ---------------------------------------------------------------------------
// Gram-Schmidt (E=8 vectors, dim 1024) + tanh combine + per-task head.
// ---------------------------------------------------------------------------
__device__ __forceinline__ float blockReduceSum(float v, float* sRed) {
#pragma unroll
    for (int off = 16; off > 0; off >>= 1) v += __shfl_xor(v, off, 32);
    const int lane = threadIdx.x & 31;
    __syncthreads();                 // guard sRed reuse across calls
    if (lane == 0) sRed[threadIdx.x >> 5] = v;
    __syncthreads();
    float r = (lane < 8) ? sRed[lane] : 0.0f;
#pragma unroll
    for (int off = 4; off > 0; off >>= 1) r += __shfl_xor(r, off, 32);
    return __shfl(r, 0, 32);
}

__global__ __launch_bounds__(256) void ortho_head_kernel(
    const float* __restrict__ h3,       // [B][E][W] f32
    const float* __restrict__ taskEmb,  // [B][E]
    const int* __restrict__ ti,         // [B]
    const float* __restrict__ W_head,   // [T][W][H]
    const float* __restrict__ b_head,   // [T][H]
    float* __restrict__ out) {          // [B][H]
    __shared__ float sBasis[E_EXP][W_DIM];
    __shared__ float sRed[8];

    const int b   = blockIdx.x;
    const int tid = threadIdx.x;
    const int d0  = tid * 4;
    const float* hb = h3 + (size_t)b * (E_EXP * W_DIM);

    float4 v0 = *reinterpret_cast<const float4*>(hb + d0);
    float  n2 = blockReduceSum(v0.x * v0.x + v0.y * v0.y + v0.z * v0.z + v0.w * v0.w, sRed);
    float  sc = 1.0f / (sqrtf(n2) + EPS_F);
    float4 b0 = make_float4(v0.x * sc, v0.y * sc, v0.z * sc, v0.w * sc);
    *reinterpret_cast<float4*>(&sBasis[0][d0]) = b0;
    __syncthreads();

    for (int i = 1; i < E_EXP; ++i) {
        float4 vi = *reinterpret_cast<const float4*>(hb + (size_t)i * W_DIM + d0);
        float coef[E_EXP - 1];
        for (int j = 0; j < i; ++j) {
            float4 bj = *reinterpret_cast<const float4*>(&sBasis[j][d0]);
            coef[j] = blockReduceSum(vi.x * bj.x + vi.y * bj.y + vi.z * bj.z + vi.w * bj.w, sRed);
        }
        float4 w = vi;
        for (int j = 0; j < i; ++j) {
            float4 bj = *reinterpret_cast<const float4*>(&sBasis[j][d0]);
            w.x -= coef[j] * bj.x; w.y -= coef[j] * bj.y;
            w.z -= coef[j] * bj.z; w.w -= coef[j] * bj.w;
        }
        float wn2 = blockReduceSum(w.x * w.x + w.y * w.y + w.z * w.z + w.w * w.w, sRed);
        float s2  = 1.0f / (sqrtf(wn2) + EPS_F);
        w.x *= s2; w.y *= s2; w.z *= s2; w.w *= s2;
        *reinterpret_cast<float4*>(&sBasis[i][d0]) = w;
        __syncthreads();
    }

    float te[E_EXP];
#pragma unroll
    for (int e = 0; e < E_EXP; ++e) te[e] = taskEmb[(size_t)b * E_EXP + e];

    float feat[4];
#pragma unroll
    for (int q = 0; q < 4; ++q) {
        float s = 0.0f;
#pragma unroll
        for (int e = 0; e < E_EXP; ++e) s += sBasis[e][d0 + q] * te[e];
        feat[q] = tanhf(s);
    }

    const int    tsel = ti[b];
    const float* Wh   = W_head + (size_t)tsel * (W_DIM * H_DIM);
    float ph[H_DIM];
#pragma unroll
    for (int h = 0; h < H_DIM; ++h) ph[h] = 0.0f;
#pragma unroll
    for (int q = 0; q < 4; ++q) {
        const int d = d0 + q;
#pragma unroll
        for (int h = 0; h < H_DIM; ++h) ph[h] += feat[q] * Wh[(size_t)d * H_DIM + h];
    }
#pragma unroll
    for (int h = 0; h < H_DIM; ++h) {
        float s = blockReduceSum(ph[h], sRed);
        if (tid == 0) out[(size_t)b * H_DIM + h] = s + b_head[(size_t)tsel * H_DIM + h];
    }
}

// ---------------------------------------------------------------------------
// Launcher
// ---------------------------------------------------------------------------
extern "C" void kernel_launch(void* const* d_in, const int* in_sizes, int n_in,
                              void* d_out, int out_size, void* d_ws, size_t ws_size,
                              hipStream_t stream) {
    const float* x      = (const float*)d_in[0];
    const float* W_task = (const float*)d_in[1];
    const float* W1     = (const float*)d_in[2];
    const float* b1     = (const float*)d_in[3];
    const float* W2     = (const float*)d_in[4];
    const float* b2     = (const float*)d_in[5];
    const float* W3     = (const float*)d_in[6];
    const float* b3     = (const float*)d_in[7];
    const float* W_head = (const float*)d_in[8];
    const float* b_head = (const float*)d_in[9];
    float* out = (float*)d_out;

    char*  ws  = (char*)d_ws;
    size_t off = 0;
    auto alloc = [&](size_t bytes) -> void* {
        void* p = ws + off;
        off += (bytes + 255) & ~(size_t)255;
        return p;
    };

    _Float16* featH = (_Float16*)alloc((size_t)B_SZ * OBS * 2);
    _Float16* W1h   = (_Float16*)alloc((size_t)E_EXP * OBS * W_DIM * 2);
    _Float16* W2h   = (_Float16*)alloc((size_t)E_EXP * W_DIM * W_DIM * 2);
    _Float16* W3h   = (_Float16*)alloc((size_t)E_EXP * W_DIM * W_DIM * 2);
    _Float16* h1H   = (_Float16*)alloc((size_t)B_SZ * E_EXP * W_DIM * 2);
    _Float16* h2H   = (_Float16*)alloc((size_t)B_SZ * E_EXP * W_DIM * 2);
    float*    h3F   = (float*)   alloc((size_t)B_SZ * E_EXP * W_DIM * 4);
    float*    tEmb  = (float*)   alloc((size_t)B_SZ * E_EXP * 4);
    int*      tiBuf = (int*)     alloc((size_t)B_SZ * 4);
    (void)ws_size; (void)in_sizes; (void)n_in; (void)out_size;

    pack_feat_kernel<<<(B_SZ * OBS) / 256, 256, 0, stream>>>(x, featH);
    {
        int n1 = E_EXP * OBS * W_DIM;
        int n2 = E_EXP * W_DIM * W_DIM;
        f32_to_f16_kernel<<<(n1 + 255) / 256, 256, 0, stream>>>(W1, W1h, n1);
        f32_to_f16_kernel<<<(n2 + 255) / 256, 256, 0, stream>>>(W2, W2h, n2);
        f32_to_f16_kernel<<<(n2 + 255) / 256, 256, 0, stream>>>(W3, W3h, n2);
    }
    task_kernel<<<B_SZ / 256, 256, 0, stream>>>(x, W_task, tEmb, tiBuf);

    dim3 gGemm(W_DIM / 64, B_SZ / 128, E_EXP);
    moe_gemm_kernel<true, _Float16><<<gGemm, 256, 0, stream>>>(
        featH, /*aExpOff=*/0, /*aStride=*/OBS, W1h, b1, h1H, /*K=*/OBS);
    moe_gemm_kernel<true, _Float16><<<gGemm, 256, 0, stream>>>(
        h1H, /*aExpOff=*/W_DIM, /*aStride=*/E_EXP * W_DIM, W2h, b2, h2H, /*K=*/W_DIM);
    moe_gemm_kernel<false, float><<<gGemm, 256, 0, stream>>>(
        h2H, /*aExpOff=*/W_DIM, /*aStride=*/E_EXP * W_DIM, W3h, b3, h3F, /*K=*/W_DIM);

    ortho_head_kernel<<<B_SZ, 256, 0, stream>>>(h3F, tEmb, tiBuf, W_head, b_head, out);
}
